// LRUModel_46935402611375
// MI455X (gfx1250) — matrix operand order repaired and checked by reading, not compiled, not run on hardware
//
#include <hip/hip_runtime.h>
#include <math.h>

// ---------------- problem constants (from reference) ----------------
#define B_   8
#define L_   8192
#define D_   256
#define HD_  512
#define DFF_ 1024
#define NL_  2
#define M_   (B_*L_)          // 65536 rows (tokens)

#define CL_  256              // scan chunk length
#define NC_  (L_/CL_)         // 32 chunks per sequence

// ---------------- WMMA types ----------------
typedef __attribute__((ext_vector_type(16))) __bf16 v16bf;
typedef __attribute__((ext_vector_type(8)))  float  v8f;

union U16B { uint4 q[2]; v16bf v; };

__device__ __forceinline__ unsigned short f2bf(float f) {
  unsigned int u = __float_as_uint(f);
  u += 0x7FFFu + ((u >> 16) & 1u);         // round-to-nearest-even
  return (unsigned short)(u >> 16);
}

// ---------------- CDNA5 async global->LDS copy (ASYNCcnt-tracked) ----------------
// 16B per lane: vdst = LDS byte offset (low 32 bits of generic shared ptr),
// vaddr = 64-bit global address, GV mode ("off").
__device__ __forceinline__ void async_ld16(void* lds_ptr, const void* gptr) {
  unsigned lo = (unsigned)(uintptr_t)lds_ptr;   // flat[31:0] == wave-relative LDS addr
  asm volatile("global_load_async_to_lds_b128 %0, %1, off"
               :: "v"(lo), "v"((unsigned long long)(uintptr_t)gptr)
               : "memory");
}
__device__ __forceinline__ void wait_async_le4() {
  asm volatile("s_wait_asynccnt 0x4" ::: "memory");
}
__device__ __forceinline__ void wait_async_0() {
  asm volatile("s_wait_asynccnt 0x0" ::: "memory");
}

// ---------------- f32 -> bf16 bulk convert ----------------
__global__ void k_f32_bf16(const float* __restrict__ s,
                           unsigned short* __restrict__ d, long n) {
  long i = ((long)blockIdx.x * blockDim.x + threadIdx.x) * 4;
  if (i + 3 < n) {
    float4 v = *(const float4*)(s + i);
    ushort4 o;
    o.x = f2bf(v.x); o.y = f2bf(v.y); o.z = f2bf(v.z); o.w = f2bf(v.w);
    *(ushort4*)(d + i) = o;
  }
}

// ---------------- per-layer parameter prep ----------------
// params_log: (NL, 3, HD) = stacked [nu_log, theta_log, gamma_log]
__global__ void k_prep(const float* __restrict__ params_log,
                       const float* __restrict__ bin_r,
                       const float* __restrict__ bin_i, int layer,
                       float* __restrict__ lam_r, float* __restrict__ lam_i,
                       float* __restrict__ lamN_r, float* __restrict__ lamN_i,
                       float* __restrict__ scale2, float* __restrict__ bias2) {
  int hd = blockIdx.x * blockDim.x + threadIdx.x;
  if (hd >= HD_) return;
  float nu = expf(params_log[(layer*3 + 0)*HD_ + hd]);
  float th = expf(params_log[(layer*3 + 1)*HD_ + hd]);
  float gm = expf(params_log[(layer*3 + 2)*HD_ + hd]);
  float mag = expf(-nu);
  float lr = mag * cosf(th), li = mag * sinf(th);
  lam_r[hd] = lr; lam_i[hd] = li;
  float ar = lr, ai = li;                      // lam^256 by squaring
  for (int s = 0; s < 8; ++s) { float nr = ar*ar - ai*ai, ni = 2.f*ar*ai; ar = nr; ai = ni; }
  lamN_r[hd] = ar; lamN_i[hd] = ai;
  scale2[hd] = gm; scale2[hd + HD_] = gm;      // h = (x@win^T + bin)*gamma
  bias2[hd]        = bin_r[layer*HD_ + hd] * gm;
  bias2[hd + HD_]  = bin_i[layer*HD_ + hd] * gm;
}

// pack win:  rows 0..511 = win_r[layer], rows 512..1023 = win_i[layer], K = D_
__global__ void k_pack_win(const float* __restrict__ wr, const float* __restrict__ wi,
                           unsigned short* __restrict__ dst, int layer) {
  int idx = blockIdx.x * blockDim.x + threadIdx.x;
  if (idx >= 2*HD_*D_) return;
  int n = idx >> 8, k = idx & (D_ - 1);
  float v = (n < HD_) ? wr[((long)layer*HD_ + n)*D_ + k]
                      : wi[((long)layer*HD_ + (n - HD_))*D_ + k];
  dst[idx] = f2bf(v);
}

// pack wout: real(h@wout^T) = h_r@wout_r^T - h_i@wout_i^T; concat on K, negate imag half
__global__ void k_pack_wout(const float* __restrict__ wr, const float* __restrict__ wi,
                            unsigned short* __restrict__ dst, int layer) {
  int idx = blockIdx.x * blockDim.x + threadIdx.x;
  if (idx >= D_*2*HD_) return;
  int n = idx >> 10, k = idx & (2*HD_ - 1);
  float v = (k < HD_) ?  wr[((long)layer*D_ + n)*HD_ + k]
                      : -wi[((long)layer*D_ + n)*HD_ + (k - HD_)];
  dst[idx] = f2bf(v);
}

// ---------------- WMMA bf16 GEMM:  C[m,n] = sum_k A[m,k]*W[n,k] + epilogue ----------------
// Block tile 128x128, 8 waves (2x4), wave tile 64x32 = 4x2 WMMA 16x16x32 tiles.
// Double-buffered LDS, async global->LDS staging overlapped with WMMA.
#define BM 128
#define BN 128
#define BK 32
#define PADK 40   // 80B row stride: 16B-aligned, bank-skewed

#define EPI_SCALEBIAS 1   // v*scale[n] + bias[n]
#define EPI_BIAS_RES  2   // v + bias[n] + res[m,n]
#define EPI_GELU      3   // gelu(v + bias[n])  (exact erf gelu)

template <int EPI>
__global__ __launch_bounds__(256)
void k_gemm_bf16(const unsigned short* __restrict__ A,   // M x K bf16
                 const unsigned short* __restrict__ W,   // N x K bf16
                 float* __restrict__ C, int M, int N, int K,
                 const float* __restrict__ e_scale,
                 const float* __restrict__ e_bias,
                 const float* __restrict__ e_res) {
  __shared__ __align__(16) unsigned short Abuf[2][BM * PADK];
  __shared__ __align__(16) unsigned short Bbuf[2][BN * PADK];

  const int tid  = threadIdx.x;
  const int wave = tid >> 5;
  const int lane = tid & 31;
  const int wm   = wave >> 2;          // 0..1
  const int wn   = wave & 3;           // 0..3
  const int m0   = blockIdx.y * BM;
  const int n0   = blockIdx.x * BN;
  const int lm   = lane & 15;
  const int kg   = lane >> 4;

  // staging geometry: A/B tiles are 128x32 bf16 = 512 x 16B chunks; 256 thr x 2
  const int srow0 = tid >> 2,          sseg0 = tid & 3;
  const int srow1 = (tid + 256) >> 2,  sseg1 = (tid + 256) & 3;

  auto stage = [&](int buf, int k0) {
    async_ld16(&Abuf[buf][srow0 * PADK + sseg0 * 8],
               A + (size_t)(m0 + srow0) * K + k0 + sseg0 * 8);
    async_ld16(&Bbuf[buf][srow0 * PADK + sseg0 * 8],
               W + (size_t)(n0 + srow0) * K + k0 + sseg0 * 8);
    async_ld16(&Abuf[buf][srow1 * PADK + sseg1 * 8],
               A + (size_t)(m0 + srow1) * K + k0 + sseg1 * 8);
    async_ld16(&Bbuf[buf][srow1 * PADK + sseg1 * 8],
               W + (size_t)(n0 + srow1) * K + k0 + sseg1 * 8);
  };

  v8f acc[4][2];
  for (int i = 0; i < 4; ++i)
    for (int j = 0; j < 2; ++j)
      for (int e = 0; e < 8; ++e) acc[i][j][e] = 0.0f;

  const int NT = K / BK;
  stage(0, 0);
  for (int it = 0; it < NT; ++it) {
    const int cur = it & 1;
    if (it + 1 < NT) {
      stage(cur ^ 1, (it + 1) * BK);   // overlap next tile's copy with this tile's math
      wait_async_le4();                // async loads retire in order: current tile resident
    } else {
      wait_async_0();
    }
    __syncthreads();

    // A frag (16x32 bf16): lane half kg -> elems 0..7: K=kg*8+e ; 8..15: K=16+kg*8+(e-8)
    v16bf afrag[4];
    for (int sm = 0; sm < 4; ++sm) {
      int mr = wm * 64 + sm * 16 + lm;
      U16B u;
      u.q[0] = *(const uint4*)&Abuf[cur][mr * PADK + kg * 8];
      u.q[1] = *(const uint4*)&Abuf[cur][mr * PADK + 16 + kg * 8];
      afrag[sm] = u.v;
    }
    // B frag (32x16 bf16): lane: column N=lm, elems 0..7: K=kg*8+e ; 8..15: K=16+kg*8+(e-8)
    v16bf bfrag[2];
    for (int sn = 0; sn < 2; ++sn) {
      int nr = wn * 32 + sn * 16 + lm;
      U16B u;
      u.q[0] = *(const uint4*)&Bbuf[cur][nr * PADK + kg * 8];
      u.q[1] = *(const uint4*)&Bbuf[cur][nr * PADK + 16 + kg * 8];
      bfrag[sn] = u.v;
    }
    for (int sm = 0; sm < 4; ++sm)
      for (int sn = 0; sn < 2; ++sn)
        acc[sm][sn] = __builtin_amdgcn_wmma_f32_16x16x32_bf16(
            false, afrag[sm], false, bfrag[sn], (short)0, acc[sm][sn], false, false);
    __syncthreads();   // all waves done reading buf[cur] before it is re-staged
  }

  // epilogue + store.  C/D layout: lane -> n=lm, vgpr r -> m = r + (lane>>4)*8
  for (int sm = 0; sm < 4; ++sm) {
    for (int sn = 0; sn < 2; ++sn) {
      int n = n0 + wn * 32 + sn * 16 + lm;
      float sc = 0.0f, bs = 0.0f;
      if constexpr (EPI == EPI_SCALEBIAS) { sc = e_scale[n]; bs = e_bias[n]; }
      else                                { bs = e_bias ? e_bias[n] : 0.0f; }
      for (int r = 0; r < 8; ++r) {
        int m = m0 + wm * 64 + sm * 16 + kg * 8 + r;
        float v = acc[sm][sn][r];
        if constexpr (EPI == EPI_SCALEBIAS) {
          v = v * sc + bs;
        } else if constexpr (EPI == EPI_BIAS_RES) {
          v = v + bs + e_res[(size_t)m * N + n];
        } else if constexpr (EPI == EPI_GELU) {
          v += bs;
          v = 0.5f * v * (1.0f + erff(v * 0.70710678118654752f));
        }
        C[(size_t)m * N + n] = v;
      }
    }
  }
}

// ---------------- LRU scan: h[l] = lam*h[l-1]*mask[l-1] + h[l]  (complex, per (b,hd)) ----
// h layout: [M_ rows][2*HD_]  (cols 0..511 = real, 512..1023 = imag)

__global__ void k_scan_local(float* __restrict__ h,
                             const unsigned char* __restrict__ mask,
                             const float* __restrict__ lam_r,
                             const float* __restrict__ lam_i,
                             float* __restrict__ carry) {
  int idx = blockIdx.x * blockDim.x + threadIdx.x;  // B_*HD_*NC_
  if (idx >= B_ * HD_ * NC_) return;
  int hd = idx & (HD_ - 1);
  int rest = idx >> 9;
  int c = rest & (NC_ - 1);
  int b = rest >> 5;
  float lr = lam_r[hd], li = lam_i[hd];
  float sr = 0.f, si = 0.f;
  long base = ((long)b * L_ + (long)c * CL_) * (2 * HD_) + hd;
  const unsigned char* mrow = mask + (long)b * L_ + (long)c * CL_;
  for (int j = 0; j < CL_; ++j) {
    float hr = h[base], hi = h[base + HD_];
    float g = (j > 0) ? (float)mrow[j - 1] : 0.0f;   // j==0: state is 0 anyway
    float cr = lr * sr - li * si;
    float ci = lr * si + li * sr;
    sr = fmaf(g, cr, hr);
    si = fmaf(g, ci, hi);
    h[base] = sr; h[base + HD_] = si;
    base += 2 * HD_;
  }
  long co = ((long)(b * NC_ + c)) * (2 * HD_) + hd;
  carry[co] = sr; carry[co + HD_] = si;
}

__global__ void k_scan_carry(const float* __restrict__ carry,
                             float* __restrict__ prefix,
                             const float* __restrict__ lamN_r,
                             const float* __restrict__ lamN_i) {
  int idx = blockIdx.x * blockDim.x + threadIdx.x;  // B_*HD_
  if (idx >= B_ * HD_) return;
  int hd = idx & (HD_ - 1);
  int b = idx >> 9;
  float Lr = lamN_r[hd], Li = lamN_i[hd];
  float sr = 0.f, si = 0.f;
  for (int c = 0; c < NC_; ++c) {
    long o = ((long)(b * NC_ + c)) * (2 * HD_) + hd;
    prefix[o] = sr; prefix[o + HD_] = si;
    float cr = carry[o], ci = carry[o + HD_];
    float tr = Lr * sr - Li * si;
    float ti = Lr * si + Li * sr;
    sr = tr + cr; si = ti + ci;
  }
}

__global__ void k_scan_fix(float* __restrict__ h,
                           const float* __restrict__ prefix,
                           const unsigned char* __restrict__ mask,
                           const float* __restrict__ lam_r,
                           const float* __restrict__ lam_i) {
  int idx = blockIdx.x * blockDim.x + threadIdx.x;  // B_*HD_*NC_
  if (idx >= B_ * HD_ * NC_) return;
  int hd = idx & (HD_ - 1);
  int rest = idx >> 9;
  int c = rest & (NC_ - 1);
  int b = rest >> 5;
  if (c == 0) return;
  long po = ((long)(b * NC_ + c)) * (2 * HD_) + hd;
  float g = (float)mask[(long)b * L_ + (long)c * CL_ - 1];
  float pr = prefix[po] * g, pi = prefix[po + HD_] * g;
  float lr = lam_r[hd], li = lam_i[hd];
  float cr = lr * pr - li * pi, ci = lr * pi + li * pr;   // lam * P
  long base = ((long)b * L_ + (long)c * CL_) * (2 * HD_) + hd;
  for (int j = 0; j < CL_; ++j) {
    h[base] += cr; h[base + HD_] += ci;
    float nr = lr * cr - li * ci, ni = lr * ci + li * cr;
    cr = nr; ci = ni;
    base += 2 * HD_;
  }
}

// ---------------- LayerNorm over D_=256, one wave32 per row ----------------
__global__ __launch_bounds__(256)
void k_layernorm(const float* __restrict__ y, const float* __restrict__ w,
                 const float* __restrict__ b, float* __restrict__ z, int M) {
  int wave = threadIdx.x >> 5, lane = threadIdx.x & 31;
  int row = blockIdx.x * 8 + wave;
  if (row >= M) return;
  const float* p = y + (size_t)row * D_ + lane * 8;
  float4 v0 = *(const float4*)p;
  float4 v1 = *(const float4*)(p + 4);
  float s = v0.x + v0.y + v0.z + v0.w + v1.x + v1.y + v1.z + v1.w;
  float q = v0.x*v0.x + v0.y*v0.y + v0.z*v0.z + v0.w*v0.w
          + v1.x*v1.x + v1.y*v1.y + v1.z*v1.z + v1.w*v1.w;
  for (int o = 16; o; o >>= 1) { s += __shfl_xor(s, o); q += __shfl_xor(q, o); }
  float mu  = s * (1.0f / D_);
  float var = q * (1.0f / D_) - mu * mu;
  float inv = rsqrtf(var + 1e-5f);
  const float* wp = w + lane * 8;
  const float* bp = b + lane * 8;
  float* zp = z + (size_t)row * D_ + lane * 8;
  float4 o0, o1;
  o0.x = (v0.x - mu) * inv * wp[0] + bp[0];
  o0.y = (v0.y - mu) * inv * wp[1] + bp[1];
  o0.z = (v0.z - mu) * inv * wp[2] + bp[2];
  o0.w = (v0.w - mu) * inv * wp[3] + bp[3];
  o1.x = (v1.x - mu) * inv * wp[4] + bp[4];
  o1.y = (v1.y - mu) * inv * wp[5] + bp[5];
  o1.z = (v1.z - mu) * inv * wp[6] + bp[6];
  o1.w = (v1.w - mu) * inv * wp[7] + bp[7];
  *(float4*)zp       = o0;
  *(float4*)(zp + 4) = o1;
}

// ---------------- host orchestration ----------------
extern "C" void kernel_launch(void* const* d_in, const int* in_sizes, int n_in,
                              void* d_out, int out_size, void* d_ws, size_t ws_size,
                              hipStream_t stream) {
  const float* x          = (const float*)d_in[0];
  const unsigned char* mask = (const unsigned char*)d_in[1];
  const float* params_log = (const float*)d_in[2];
  const float* win_r      = (const float*)d_in[3];
  const float* win_i      = (const float*)d_in[4];
  const float* bin_r      = (const float*)d_in[5];
  const float* bin_i      = (const float*)d_in[6];
  const float* wout_r     = (const float*)d_in[7];
  const float* wout_i     = (const float*)d_in[8];
  const float* bout_r     = (const float*)d_in[9];
  // bout_i (d_in[10]) only affects imag part, dropped by real()
  const float* ln1_w      = (const float*)d_in[11];
  const float* ln1_b      = (const float*)d_in[12];
  const float* w1         = (const float*)d_in[13];
  const float* b1         = (const float*)d_in[14];
  const float* w2         = (const float*)d_in[15];
  const float* b2         = (const float*)d_in[16];
  const float* ln2_w      = (const float*)d_in[17];
  const float* ln2_b      = (const float*)d_in[18];

  char* ws = (char*)d_ws;
  size_t off = 0;
  auto alloc = [&](size_t bytes) -> char* {
    char* p = ws + off;
    off += (bytes + 255) & ~(size_t)255;
    return p;
  };
  float*          hbuf   = (float*)alloc((size_t)M_ * 2 * HD_ * 4);   // also FFN hidden
  unsigned short* abf    = (unsigned short*)alloc((size_t)M_ * 2 * HD_ * 2);
  float*          ybuf   = (float*)alloc((size_t)M_ * D_ * 4);
  float*          zbuf   = (float*)alloc((size_t)M_ * D_ * 4);
  float*          outbuf = (float*)alloc((size_t)M_ * D_ * 4);
  unsigned short* winc   = (unsigned short*)alloc((size_t)2 * HD_ * D_ * 2);
  unsigned short* woutc  = (unsigned short*)alloc((size_t)D_ * 2 * HD_ * 2);
  unsigned short* w1b    = (unsigned short*)alloc((size_t)DFF_ * D_ * 2);
  unsigned short* w2b    = (unsigned short*)alloc((size_t)D_ * DFF_ * 2);
  float* lam_r  = (float*)alloc(HD_ * 4);
  float* lam_i  = (float*)alloc(HD_ * 4);
  float* lamN_r = (float*)alloc(HD_ * 4);
  float* lamN_i = (float*)alloc(HD_ * 4);
  float* scale2 = (float*)alloc(2 * HD_ * 4);
  float* bias2  = (float*)alloc(2 * HD_ * 4);
  float* carry  = (float*)alloc((size_t)B_ * NC_ * 2 * HD_ * 4);
  float* prefix = (float*)alloc((size_t)B_ * NC_ * 2 * HD_ * 4);

  auto conv = [&](const float* s, unsigned short* d, long n) {
    k_f32_bf16<<<(unsigned)((n / 4 + 255) / 256), 256, 0, stream>>>(s, d, n);
  };

  for (int layer = 0; layer < NL_; ++layer) {
    const float* lin  = (layer == 0) ? x : outbuf;
    float*       lout = (layer == NL_ - 1) ? (float*)d_out : outbuf;

    // per-layer param/weight prep
    k_prep<<<2, 256, 0, stream>>>(params_log, bin_r, bin_i, layer,
                                  lam_r, lam_i, lamN_r, lamN_i, scale2, bias2);
    k_pack_win<<<(2*HD_*D_ + 255)/256, 256, 0, stream>>>(win_r, win_i, winc, layer);
    k_pack_wout<<<(D_*2*HD_ + 255)/256, 256, 0, stream>>>(wout_r, wout_i, woutc, layer);
    conv(w1 + (size_t)layer * DFF_ * D_, w1b, (long)DFF_ * D_);
    conv(w2 + (size_t)layer * D_ * DFF_, w2b, (long)D_ * DFF_);

    // input projection (complex): h = (x @ win^T + bin) * gamma
    conv(lin, abf, (long)M_ * D_);
    k_gemm_bf16<EPI_SCALEBIAS><<<dim3(2*HD_/BN, M_/BM), 256, 0, stream>>>(
        abf, winc, hbuf, M_, 2*HD_, D_, scale2, bias2, nullptr);

    // LRU scan (3-phase chunked)
    k_scan_local<<<(B_*HD_*NC_ + 255)/256, 256, 0, stream>>>(hbuf, mask, lam_r, lam_i, carry);
    k_scan_carry<<<(B_*HD_ + 255)/256, 256, 0, stream>>>(carry, prefix, lamN_r, lamN_i);
    k_scan_fix<<<(B_*HD_*NC_ + 255)/256, 256, 0, stream>>>(hbuf, prefix, mask, lam_r, lam_i);

    // output projection: y = real(h @ wout^T) + bout_r + residual
    conv(hbuf, abf, (long)M_ * 2 * HD_);
    k_gemm_bf16<EPI_BIAS_RES><<<dim3(D_/BN, M_/BM), 256, 0, stream>>>(
        abf, woutc, ybuf, M_, D_, 2*HD_, nullptr, bout_r + (size_t)layer * D_, lin);

    // LN1
    k_layernorm<<<M_ / 8, 256, 0, stream>>>(ybuf, ln1_w + (size_t)layer * D_,
                                            ln1_b + (size_t)layer * D_, zbuf, M_);

    // FFN: gelu(z @ w1^T + b1) @ w2^T + b2 + z
    conv(zbuf, abf, (long)M_ * D_);
    k_gemm_bf16<EPI_GELU><<<dim3(DFF_/BN, M_/BM), 256, 0, stream>>>(
        abf, w1b, hbuf, M_, DFF_, D_, nullptr, b1 + (size_t)layer * DFF_, nullptr);
    conv(hbuf, abf, (long)M_ * DFF_);
    k_gemm_bf16<EPI_BIAS_RES><<<dim3(D_/BN, M_/BM), 256, 0, stream>>>(
        abf, w2b, ybuf, M_, D_, DFF_, nullptr, b2 + (size_t)layer * D_, zbuf);

    // LN2 -> layer output
    k_layernorm<<<M_ / 8, 256, 0, stream>>>(ybuf, ln2_w + (size_t)layer * D_,
                                            ln2_b + (size_t)layer * D_, lout, M_);
  }
}
